// RelPosQTransformerLayer_25374666784995
// MI455X (gfx1250) — matrix operand order repaired and checked by reading, not compiled, not run on hardware
//
#include <hip/hip_runtime.h>
#include <cmath>

// ---- problem constants (match reference) ----
constexpr int cB = 2, cQ = 512, cA = 4096, cD = 512, cH = 8, cDFF = 2048, cMD = 100;
constexpr int cDK = cD / cH;   // 64
constexpr float cNEG = -1e9f;

typedef __attribute__((ext_vector_type(16))) __bf16 v16bf;
typedef __attribute__((ext_vector_type(8)))  float  v8f;

__device__ __forceinline__ v8f zero_v8f() {
  v8f z;
#pragma unroll
  for (int i = 0; i < 8; ++i) z[i] = 0.0f;
  return z;
}

// ---- CDNA5 async global->LDS DMA (ASYNCcnt path, CDNA5 ISA) ----
// VGLOBAL encoding, GV mode: vdst = LDS byte address, vaddr = 64-bit global addr.
__device__ __forceinline__ void async_load_b128(unsigned int lds_byte, const void* gptr) {
  asm volatile("global_load_async_to_lds_b128 %0, %1, off"
               :: "v"(lds_byte), "v"((unsigned long long)(uintptr_t)gptr)
               : "memory");
}
// Threshold wait: ASYNCcnt <= n (async loads complete in order -> older tile drained)
__device__ __forceinline__ void wait_async_le(int n) {
  if (n == 0)      asm volatile("s_wait_asynccnt 0x0"  ::: "memory");
  else if (n == 4) asm volatile("s_wait_asynccnt 0x4"  ::: "memory");
  else             asm volatile("s_wait_asynccnt 0x10" ::: "memory");
}
// Generic (flat) shared pointer -> raw LDS byte offset (flat->LDS keeps addr[31:0])
__device__ __forceinline__ unsigned int lds_off(const void* p) {
  return (unsigned int)(uintptr_t)p;
}

// ---- WMMA fragment loaders (contiguous k order k = 16*(lane/16) + e, same for A and B, wave32) ----
// A matrix 16x32 bf16: m = lane&15; lanes 0-15 hold K 0-15, lanes 16-31 hold K 16-31
__device__ __forceinline__ v16bf load_a_frag(const __bf16* base, int ld, int lane) {
  v16bf a;
  const int m  = lane & 15;
  const int kb = (lane >> 4) << 4;
#pragma unroll
  for (int e = 0; e < 16; ++e) {
    const int k = kb + e;
    a[e] = base[m * ld + k];
  }
  return a;
}
// B matrix 32x16 bf16 from [n][k]-major tile: n = lane&15; lanes 0-15 K=0-15, lanes 16-31 K=16-31
__device__ __forceinline__ v16bf load_b_frag_nk(const __bf16* base, int ld, int lane) {
  v16bf bf;
  const int n  = lane & 15;
  const int kb = (lane >> 4) << 4;
#pragma unroll
  for (int e = 0; e < 16; ++e) bf[e] = base[n * ld + kb + e];
  return bf;
}
// B matrix 32x16 bf16 from [k][n]-major tile (e.g. V chunk)
__device__ __forceinline__ v16bf load_b_frag_kn(const __bf16* base, int ld, int lane) {
  v16bf bf;
  const int n  = lane & 15;
  const int kb = (lane >> 4) << 4;
#pragma unroll
  for (int e = 0; e < 16; ++e) bf[e] = base[(kb + e) * ld + n];
  return bf;
}

// ============================================================================
// fp32 -> bf16 elementwise (one-time precision demotion; halves HBM traffic)
// ============================================================================
__global__ __launch_bounds__(256)
void cvt_f32_to_bf16(const float* __restrict__ in, __bf16* __restrict__ out, int n) {
  const int i4 = (blockIdx.x * 256 + threadIdx.x) * 4;
  if (i4 + 3 < n) {
    const float4 f = *(const float4*)(in + i4);
    out[i4 + 0] = (__bf16)f.x; out[i4 + 1] = (__bf16)f.y;
    out[i4 + 2] = (__bf16)f.z; out[i4 + 3] = (__bf16)f.w;
  }
}

// ============================================================================
// GEMM: C[M,N] = act(X[M,K] @ W[N,K]^T + bias[N]).  bf16 in, f32 accumulate,
// optional fp32 and/or bf16 outputs.  Block = 128 threads (4 waves), 64x64
// tile, 4 WMMAs per wave per 32-K step.  Double-buffered async global->LDS
// DMA: tile k+1 streams while tile k is consumed (wait ASYNCcnt <= 4).
// ============================================================================
__global__ __launch_bounds__(128)
void wmma_gemm_bf16(const __bf16* __restrict__ X, const __bf16* __restrict__ W,
                    const float* __restrict__ bias,
                    float* __restrict__ Cf, __bf16* __restrict__ Cb,
                    int M, int N, int K, int relu)
{
  __shared__ __bf16 sA[2][64 * 32];
  __shared__ __bf16 sB[2][64 * 32];
  const int tid  = threadIdx.x;
  const int lane = tid & 31;
  const int wave = tid >> 5;
  const int m0 = blockIdx.x * 64;
  const int n0 = blockIdx.y * 64;
  const unsigned int ldsA[2] = { lds_off(sA[0]), lds_off(sA[1]) };
  const unsigned int ldsB[2] = { lds_off(sB[0]), lds_off(sB[1]) };

  v8f acc[4];
#pragma unroll
  for (int t = 0; t < 4; ++t) acc[t] = zero_v8f();

  auto stage = [&](int k0, int buf) {   // 4 async b128 ops per thread
#pragma unroll
    for (int i = 0; i < 2; ++i) {
      const int idx = tid + i * 128;    // 256 x 16B per matrix tile
      const int row = idx >> 2, c = idx & 3;
      const unsigned int lofs = (unsigned int)(row * 32 + c * 8) * 2;
      async_load_b128(ldsA[buf] + lofs, X + (size_t)(m0 + row) * K + k0 + c * 8);
      async_load_b128(ldsB[buf] + lofs, W + (size_t)(n0 + row) * K + k0 + c * 8);
    }
  };

  stage(0, 0);
  for (int k0 = 0; k0 < K; k0 += 32) {
    const int cur = (k0 >> 5) & 1;
    const bool more = (k0 + 32 < K);
    if (more) stage(k0 + 32, cur ^ 1);             // next tile DMA in flight
    if (k0 + 64 < K)                               // L2 prefetch two tiles out
      __builtin_prefetch(W + (size_t)(n0 + (tid >> 1)) * K + k0 + 64, 0, 1);
    wait_async_le(more ? 4 : 0);                   // drain current tile only
    __syncthreads();
    const v16bf a = load_a_frag(&sA[cur][(wave * 16) * 32], 32, lane);
#pragma unroll
    for (int t = 0; t < 4; ++t) {
      v16bf b = load_b_frag_nk(&sB[cur][(t * 16) * 32], 32, lane);
      acc[t] = __builtin_amdgcn_wmma_f32_16x16x32_bf16(false, a, false, b, (short)0,
                                                       acc[t], false, false);
    }
    __syncthreads();
  }

  const int mb  = (lane >> 4) << 3;
  const int n16 = lane & 15;
#pragma unroll
  for (int t = 0; t < 4; ++t) {
    const int n = n0 + t * 16 + n16;
    const float bn = bias[n];
#pragma unroll
    for (int v = 0; v < 8; ++v) {
      float r = acc[t][v] + bn;
      if (relu) r = fmaxf(r, 0.0f);
      const size_t o = (size_t)(m0 + wave * 16 + mb + v) * N + n;
      if (Cf) Cf[o] = r;
      if (Cb) Cb[o] = (__bf16)r;
    }
  }
}

// ============================================================================
// Flash-style attention with relative-position bias (bf16 Q/K/V, f32 softmax).
// One wave per (b, h, 16-query tile); keys streamed in chunks of 32 with
// double-buffered async global->LDS DMA (16 ops/chunk, wait ASYNCcnt <= 16).
// ============================================================================
__global__ __launch_bounds__(32)
void attn_wmma_kernel(const __bf16* __restrict__ Qp, const __bf16* __restrict__ Kp,
                      const __bf16* __restrict__ Vp,
                      const int* __restrict__ qx, const int* __restrict__ qy,
                      const unsigned char* __restrict__ qpm,
                      const int* __restrict__ axp, const int* __restrict__ ayp,
                      const unsigned char* __restrict__ apm,
                      const unsigned char* __restrict__ fullm,
                      const unsigned char* __restrict__ padm,
                      const float* __restrict__ pex, const float* __restrict__ pey,
                      __bf16* __restrict__ ctx)
{
  __shared__ __bf16 sQ[16 * 64];
  __shared__ __bf16 sK[2][32 * 64];
  __shared__ __bf16 sV[2][32 * 64];
  __shared__ float  sS[16 * 32];
  __shared__ __bf16 sP[16 * 32];
  __shared__ float  sScale[16];
  __shared__ int    sQX[16], sQY[16], sAX[32], sAY[32];
  __shared__ unsigned char sQPM[16], sAPM[32], sPad[32];

  const int lane = threadIdx.x;
  const int nQT = cQ / 16;
  const int bid = blockIdx.x;
  const int qt = bid % nQT;
  const int h  = (bid / nQT) % cH;
  const int b  = bid / (nQT * cH);
  const int q0 = qt * 16;
  const unsigned int ldsQ = lds_off(sQ);
  const unsigned int ldsK[2] = { lds_off(sK[0]), lds_off(sK[1]) };
  const unsigned int ldsV[2] = { lds_off(sV[0]), lds_off(sV[1]) };

  auto stage_kv = [&](int a0, int buf) {   // 16 async b128 ops per lane
#pragma unroll
    for (int i = 0; i < 8; ++i) {
      const int idx = lane + i * 32;       // 256 x 16B per matrix
      const int row = idx >> 3, c = idx & 7;
      const size_t g = (size_t)(b * cA + a0 + row) * cD + h * cDK + c * 8;
      const unsigned int lofs = (unsigned int)(row * 64 + c * 8) * 2;
      async_load_b128(ldsK[buf] + lofs, Kp + g);
      async_load_b128(ldsV[buf] + lofs, Vp + g);
    }
  };

  // prologue: Q tile DMA (4 ops) overlapped with chunk-0 K/V DMA (16 ops)
#pragma unroll
  for (int i = 0; i < 4; ++i) {
    const int idx = lane + i * 32;
    const int row = idx >> 3, c = idx & 7;
    async_load_b128(ldsQ + (unsigned int)(row * 64 + c * 8) * 2,
                    Qp + (size_t)(b * cQ + q0 + row) * cD + h * cDK + c * 8);
  }
  stage_kv(0, 0);
  if (lane < 16) {
    sQX[lane]  = qx[b * cQ + q0 + lane];
    sQY[lane]  = qy[b * cQ + q0 + lane];
    sQPM[lane] = qpm[b * cQ + q0 + lane];
  }
  wait_async_le(16);                       // Q resident; chunk 0 still in flight
  __syncthreads();

  const v16bf aQ0 = load_a_frag(&sQ[0],  64, lane);   // dk 0..31
  const v16bf aQ1 = load_a_frag(&sQ[32], 64, lane);   // dk 32..63

  v8f acc[4];
  for (int t = 0; t < 4; ++t) acc[t] = zero_v8f();
  float rowM = -3.0e38f, rowL = 0.0f;

  const int mb  = (lane >> 4) << 3;   // C layout: row = mb + vgpr index
  const int n16 = lane & 15;          // C layout: col = n16

  for (int a0 = 0; a0 < cA; a0 += 32) {
    const int cur = (a0 >> 5) & 1;
    const bool more = (a0 + 32 < cA);
    if (more) stage_kv(a0 + 32, cur ^ 1);  // next chunk DMA in flight
    sAX[lane]  = axp[b * cA + a0 + lane];
    sAY[lane]  = ayp[b * cA + a0 + lane];
    sAPM[lane] = apm[b * cA + a0 + lane];
    sPad[lane] = padm[b * cA + a0 + lane];
    wait_async_le(more ? 16 : 0);          // drain current chunk only
    __syncthreads();

    // scores = Q K^T / 8  (two 16-key N-tiles, K=64 contraction in 2 steps)
#pragma unroll
    for (int nt = 0; nt < 2; ++nt) {
      v8f sc = zero_v8f();
      v16bf b0 = load_b_frag_nk(&sK[cur][(nt * 16) * 64 + 0],  64, lane);
      sc = __builtin_amdgcn_wmma_f32_16x16x32_bf16(false, aQ0, false, b0, (short)0, sc, false, false);
      v16bf b1 = load_b_frag_nk(&sK[cur][(nt * 16) * 64 + 32], 64, lane);
      sc = __builtin_amdgcn_wmma_f32_16x16x32_bf16(false, aQ1, false, b1, (short)0, sc, false, false);
      const int kl  = nt * 16 + n16;
      const int key = a0 + kl;
#pragma unroll
      for (int v = 0; v < 8; ++v) {
        const int m = mb + v;
        float s = sc[v] * 0.125f;          // 1/sqrt(64)
        int rx = sQX[m] - sAX[kl]; rx = rx < -cMD ? -cMD : (rx > cMD ? cMD : rx); rx += cMD;
        int ry = sQY[m] - sAY[kl]; ry = ry < -cMD ? -cMD : (ry > cMD ? cMD : ry); ry += cMD;
        float bias = pex[rx * cH + h] + pey[ry * cH + h];
        if (!(sQPM[m] && sAPM[kl])) bias = 0.0f;
        s += bias;
        if (!fullm[((size_t)b * cQ + q0 + m) * cA + key]) s = cNEG;
        if (sPad[kl]) s = cNEG;
        sS[m * 32 + kl] = s;
      }
    }
    __syncthreads();

    // online softmax bookkeeping: one lane per query row
    if (lane < 16) {
      const int r = lane;
      float cmax = -3.0e38f;
      for (int j = 0; j < 32; ++j) cmax = fmaxf(cmax, sS[r * 32 + j]);
      const float mnew = fmaxf(rowM, cmax);
      const float corr = (rowM > -1.0e38f) ? __expf(rowM - mnew) : 0.0f;
      float lsum = 0.0f;
      for (int j = 0; j < 32; ++j) {
        const float p = __expf(sS[r * 32 + j] - mnew);
        sP[r * 32 + j] = (__bf16)p;
        lsum += p;
      }
      rowL = rowL * corr + lsum;
      rowM = mnew;
      sScale[r] = corr;
    }
    __syncthreads();

    float f[8];
#pragma unroll
    for (int v = 0; v < 8; ++v) f[v] = sScale[mb + v];
#pragma unroll
    for (int t = 0; t < 4; ++t)
#pragma unroll
      for (int v = 0; v < 8; ++v) acc[t][v] *= f[v];

    // ctx += P(16x32) @ V(32x64)
    const v16bf aP = load_a_frag(sP, 32, lane);
#pragma unroll
    for (int t = 0; t < 4; ++t) {
      v16bf bv = load_b_frag_kn(&sV[cur][t * 16], 64, lane);
      acc[t] = __builtin_amdgcn_wmma_f32_16x16x32_bf16(false, aP, false, bv, (short)0,
                                                       acc[t], false, false);
    }
    __syncthreads();
  }

  if (lane < 16) sScale[lane] = (rowL > 0.0f) ? 1.0f / rowL : 0.0f;
  __syncthreads();
  float f[8];
#pragma unroll
  for (int v = 0; v < 8; ++v) f[v] = sScale[mb + v];
#pragma unroll
  for (int t = 0; t < 4; ++t)
#pragma unroll
    for (int v = 0; v < 8; ++v) {
      const int m = mb + v;
      ctx[(size_t)(b * cQ + q0 + m) * cD + h * cDK + t * 16 + n16] = (__bf16)(acc[t][v] * f[v]);
    }
}

// ============================================================================
// out = LayerNorm(xa + xb) * g + beta  (row = cD = 512, 256 threads); optional
// bf16 mirror output for downstream WMMA consumers.
// ============================================================================
__global__ __launch_bounds__(256)
void add_layernorm(const float* __restrict__ xa, const float* __restrict__ xb,
                   const float* __restrict__ g, const float* __restrict__ beta,
                   float* __restrict__ outf, __bf16* __restrict__ outb)
{
  __shared__ float red[8];
  const int row = blockIdx.x, t = threadIdx.x;
  const size_t base = (size_t)row * cD;
  const float v0 = xa[base + t] + xb[base + t];
  const float v1 = xa[base + t + 256] + xb[base + t + 256];
  float s = v0 + v1;
#pragma unroll
  for (int o = 16; o > 0; o >>= 1) s += __shfl_xor(s, o, 32);
  if ((t & 31) == 0) red[t >> 5] = s;
  __syncthreads();
  float tot = 0.0f;
#pragma unroll
  for (int i = 0; i < 8; ++i) tot += red[i];
  const float mean = tot * (1.0f / cD);
  __syncthreads();
  const float d0 = v0 - mean, d1 = v1 - mean;
  float vs = d0 * d0 + d1 * d1;
#pragma unroll
  for (int o = 16; o > 0; o >>= 1) vs += __shfl_xor(vs, o, 32);
  if ((t & 31) == 0) red[t >> 5] = vs;
  __syncthreads();
  float vtot = 0.0f;
#pragma unroll
  for (int i = 0; i < 8; ++i) vtot += red[i];
  const float inv = rsqrtf(vtot * (1.0f / cD) + 1e-5f);
  const float r0 = d0 * inv * g[t]       + beta[t];
  const float r1 = d1 * inv * g[t + 256] + beta[t + 256];
  if (outf) { outf[base + t] = r0; outf[base + t + 256] = r1; }
  if (outb) { outb[base + t] = (__bf16)r0; outb[base + t + 256] = (__bf16)r1; }
}

// ============================================================================
extern "C" void kernel_launch(void* const* d_in, const int* in_sizes, int n_in,
                              void* d_out, int out_size, void* d_ws, size_t ws_size,
                              hipStream_t stream) {
  (void)in_sizes; (void)n_in; (void)out_size; (void)ws_size;
  const float* q_tok = (const float*)d_in[0];
  const int*   qx    = (const int*)d_in[1];
  const int*   qy    = (const int*)d_in[2];
  const unsigned char* qpm = (const unsigned char*)d_in[3];
  const float* a_tok = (const float*)d_in[4];
  const int*   ax    = (const int*)d_in[5];
  const int*   ay    = (const int*)d_in[6];
  const unsigned char* apm   = (const unsigned char*)d_in[7];
  const unsigned char* fullm = (const unsigned char*)d_in[8];
  const unsigned char* padm  = (const unsigned char*)d_in[9];
  const float *Wq = (const float*)d_in[10], *bq = (const float*)d_in[11];
  const float *Wk = (const float*)d_in[12], *bk = (const float*)d_in[13];
  const float *Wv = (const float*)d_in[14], *bv = (const float*)d_in[15];
  const float *Wo = (const float*)d_in[16], *bo = (const float*)d_in[17];
  const float *pex = (const float*)d_in[18], *pey = (const float*)d_in[19];
  const float *W1 = (const float*)d_in[20], *b1 = (const float*)d_in[21];
  const float *W2 = (const float*)d_in[22], *b2 = (const float*)d_in[23];
  const float *g1 = (const float*)d_in[24], *be1 = (const float*)d_in[25];
  const float *g2 = (const float*)d_in[26], *be2 = (const float*)d_in[27];

  const int MQ = cB * cQ;   // 1024
  const int MA = cB * cA;   // 8192

  // carve workspace (256B-aligned chunks)
  char* w = (char*)d_ws;
  auto alloc_bf = [&](size_t n) { __bf16* p = (__bf16*)w; w += ((n * 2 + 255) & ~(size_t)255); return p; };
  auto alloc_f  = [&](size_t n) { float*  p = (float*)w;  w += ((n * 4 + 255) & ~(size_t)255); return p; };
  __bf16 *Xq  = alloc_bf((size_t)MQ * cD);
  __bf16 *Xa  = alloc_bf((size_t)MA * cD);
  __bf16 *Wqb = alloc_bf((size_t)cD * cD);
  __bf16 *Wkb = alloc_bf((size_t)cD * cD);
  __bf16 *Wvb = alloc_bf((size_t)cD * cD);
  __bf16 *Wob = alloc_bf((size_t)cD * cD);
  __bf16 *W1b = alloc_bf((size_t)cDFF * cD);
  __bf16 *W2b = alloc_bf((size_t)cD * cDFF);
  __bf16 *Qp  = alloc_bf((size_t)MQ * cD);
  __bf16 *Kp  = alloc_bf((size_t)MA * cD);
  __bf16 *Vp  = alloc_bf((size_t)MA * cD);
  __bf16 *ctxb = alloc_bf((size_t)MQ * cD);
  __bf16 *x1b  = alloc_bf((size_t)MQ * cD);
  __bf16 *ffhb = alloc_bf((size_t)MQ * cDFF);
  float  *ao   = alloc_f((size_t)MQ * cD);
  float  *x1f  = alloc_f((size_t)MQ * cD);
  float  *ffo  = alloc_f((size_t)MQ * cD);
  float  *outp = (float*)d_out;

  auto cvt_grid = [](size_t n) { return (unsigned)((n / 4 + 255) / 256); };

  // one-time fp32 -> bf16 demotion of activations and weights
  cvt_f32_to_bf16<<<cvt_grid((size_t)MQ * cD), 256, 0, stream>>>(q_tok, Xq, MQ * cD);
  cvt_f32_to_bf16<<<cvt_grid((size_t)MA * cD), 256, 0, stream>>>(a_tok, Xa, MA * cD);
  cvt_f32_to_bf16<<<cvt_grid((size_t)cD * cD), 256, 0, stream>>>(Wq, Wqb, cD * cD);
  cvt_f32_to_bf16<<<cvt_grid((size_t)cD * cD), 256, 0, stream>>>(Wk, Wkb, cD * cD);
  cvt_f32_to_bf16<<<cvt_grid((size_t)cD * cD), 256, 0, stream>>>(Wv, Wvb, cD * cD);
  cvt_f32_to_bf16<<<cvt_grid((size_t)cD * cD), 256, 0, stream>>>(Wo, Wob, cD * cD);
  cvt_f32_to_bf16<<<cvt_grid((size_t)cDFF * cD), 256, 0, stream>>>(W1, W1b, cDFF * cD);
  cvt_f32_to_bf16<<<cvt_grid((size_t)cD * cDFF), 256, 0, stream>>>(W2, W2b, cD * cDFF);

  // Q/K/V projections (bf16 outputs only — attention consumes bf16)
  wmma_gemm_bf16<<<dim3(MQ / 64, cD / 64), 128, 0, stream>>>(Xq, Wqb, bq, nullptr, Qp, MQ, cD, cD, 0);
  wmma_gemm_bf16<<<dim3(MA / 64, cD / 64), 128, 0, stream>>>(Xa, Wkb, bk, nullptr, Kp, MA, cD, cD, 0);
  wmma_gemm_bf16<<<dim3(MA / 64, cD / 64), 128, 0, stream>>>(Xa, Wvb, bv, nullptr, Vp, MA, cD, cD, 0);

  // attention with relative position bias (flash-style online softmax)
  attn_wmma_kernel<<<cB * cH * (cQ / 16), 32, 0, stream>>>(
      Qp, Kp, Vp, qx, qy, qpm, ax, ay, apm, fullm, padm, pex, pey, ctxb);

  // output projection (fp32 for residual), residual + LN1 (fp32 + bf16 mirror)
  wmma_gemm_bf16<<<dim3(MQ / 64, cD / 64), 128, 0, stream>>>(ctxb, Wob, bo, ao, nullptr, MQ, cD, cD, 0);
  add_layernorm<<<MQ, 256, 0, stream>>>(q_tok, ao, g1, be1, x1f, x1b);

  // FFN: relu(x @ W1^T + b1) @ W2^T + b2, residual + LN2 -> out
  wmma_gemm_bf16<<<dim3(MQ / 64, cDFF / 64), 128, 0, stream>>>(x1b, W1b, b1, nullptr, ffhb, MQ, cDFF, cD, 1);
  wmma_gemm_bf16<<<dim3(MQ / 64, cD / 64), 128, 0, stream>>>(ffhb, W2b, b2, ffo, nullptr, MQ, cD, cDFF, 0);
  add_layernorm<<<MQ, 256, 0, stream>>>(x1f, ffo, g2, be2, outp, nullptr);
}
